// Attention_609885356911
// MI455X (gfx1250) — matrix operand order repaired and checked by reading, not compiled, not run on hardware
//
#include <hip/hip_runtime.h>
#include <hip/hip_bf16.h>

// ---------------------------------------------------------------------------
// SAM-style windowed ViT attention w/ decomposed rel-pos + LoRA.
// MI455X / gfx1250: wave32, v_wmma_f32_16x16x32_bf16, flash attention,
// async global->LDS staging (GLOBAL_LOAD_ASYNC_TO_LDS_B128 + ASYNCcnt)
// shared across 8-wave workgroups, double-buffered against WMMA compute.
// LoRA rank-12 update folded into WMMA via zero-padded K=32 operands.
// ---------------------------------------------------------------------------

#define NUM_HEADS 12
#define DIM 768
#define HEAD_DIM 64
#define NTOK 4096          // 64*64 tokens
#define QKV_LD 2304        // 3*DIM
#define SCALE 0.125f       // HEAD_DIM^-0.5 (exact power of two)

typedef __attribute__((ext_vector_type(16))) __bf16          v16bf;
typedef __attribute__((ext_vector_type(16))) unsigned short  v16us;
typedef __attribute__((ext_vector_type(8)))  unsigned short  v8us;
typedef __attribute__((ext_vector_type(8)))  float           v8f;

// pack two fp32 -> two bf16 (round-half-up): 2 adds + one v_perm_b32
__device__ __forceinline__ unsigned pk2bf(float lo, float hi) {
    unsigned a = __builtin_bit_cast(unsigned, lo) + 0x8000u;
    unsigned b = __builtin_bit_cast(unsigned, hi) + 0x8000u;
    return __builtin_amdgcn_perm(b, a, 0x07060302u);
}

__device__ __forceinline__ unsigned short f2bf(float f) {
    unsigned u = __builtin_bit_cast(unsigned, f) + 0x8000u;
    return (unsigned short)(u >> 16);
}

__device__ __forceinline__ v8f wmma_bf16(v16bf a, v16bf b, v8f c) {
    return __builtin_amdgcn_wmma_f32_16x16x32_bf16(
        false, a, false, b, (short)0, c, false, false);
}

// ---- CDNA5 async global->LDS copy (16B per lane), tracked by ASYNCcnt -----
__device__ __forceinline__ void async_b128(unsigned lds_byte_off,
                                           const unsigned short* g) {
    unsigned long long ga = (unsigned long long)(uintptr_t)g;
    asm volatile("global_load_async_to_lds_b128 %0, %1, off"
                 :: "v"(lds_byte_off), "v"(ga) : "memory");
}
#define WAIT_ASYNC_4() asm volatile("s_wait_asynccnt 0x4" ::: "memory")
#define WAIT_ASYNC_0() asm volatile("s_wait_asynccnt 0x0" ::: "memory")

// ---- WMMA operand loaders (bf16 sources; work for global or LDS ptrs) -----
// A (16x32): lane l -> m=l&15, hi=l>>4; k = 16*hi + s (two 16B chunks).
__device__ __forceinline__ v16bf load_a_bf(const unsigned short* p, int ld,
                                           int row0, int k0) {
    int l = threadIdx.x & 31, m = l & 15, hi = l >> 4;
    const unsigned short* r = p + (size_t)(row0 + m) * ld + k0 + hi * 16;
    v8us lo = *(const v8us*)r;
    v8us hh = *(const v8us*)(r + 8);
    v16us u = __builtin_shufflevector(lo, hh, 0, 1, 2, 3, 4, 5, 6, 7,
                                      8, 9, 10, 11, 12, 13, 14, 15);
    return __builtin_bit_cast(v16bf, u);
}

// B (32x16): lane l -> n=l&15, kg=l>>4; one 32B contiguous read of "row n"
// of a row-major [n][k] source (B = src^T).
__device__ __forceinline__ v16bf load_b_bf(const unsigned short* p, int ld,
                                           int n0, int k0) {
    int l = threadIdx.x & 31, n = l & 15, kg = l >> 4;
    v16us u = *(const v16us*)(p + (size_t)(n0 + n) * ld + k0 + kg * 16);
    return __builtin_bit_cast(v16bf, u);
}

// B operand with per-column row selection (rel-pos tables, bf16, clamped).
__device__ __forceinline__ v16bf load_b_bf_step(const unsigned short* p, int ld,
                                                int rowBase, int rowStep,
                                                int maxRow, int k0) {
    int l = threadIdx.x & 31, n = l & 15, kg = l >> 4;
    int row = rowBase + rowStep * n;
    row = row < 0 ? 0 : (row > maxRow ? maxRow : row);
    v16us u = *(const v16us*)(p + (size_t)row * ld + k0 + kg * 16);
    return __builtin_bit_cast(v16bf, u);
}

// ---------------------------------------------------------------------------
// one-shot fp32 -> packed bf16
// ---------------------------------------------------------------------------
__global__ void cvt_bf16_kernel(const float* __restrict__ src,
                                unsigned* __restrict__ dst, int npairs) {
    int i = blockIdx.x * blockDim.x + threadIdx.x;
    if (i >= npairs) return;
    float2 f = ((const float2*)src)[i];
    dst[i] = pk2bf(f.x, f.y);
}

// ---------------------------------------------------------------------------
// LoRA bottleneck, zero-padded to K=32: t32[4096][32] bf16 (cols 12..31 = 0)
// ---------------------------------------------------------------------------
__global__ void lora_t_kernel(const float* __restrict__ x,
                              const float* __restrict__ loraA,
                              unsigned short* __restrict__ t32) {
    int idx = blockIdx.x * blockDim.x + threadIdx.x;   // NTOK*32
    if (idx >= NTOK * 32) return;
    int m = idx >> 5, j = idx & 31;
    unsigned short r = 0;
    if (j < 12) {
        const float* xr = x + (size_t)m * DIM;
        const float* ar = loraA + (size_t)j * DIM;
        float s = 0.f;
#pragma unroll 8
        for (int c = 0; c < DIM; ++c) s += xr[c] * ar[c];
        r = f2bf(s);
    }
    t32[idx] = r;
}

// loraB32[2304][32] bf16 (cols 12..31 = 0)
__global__ void loraB32_kernel(const float* __restrict__ loraB,
                               unsigned short* __restrict__ o) {
    int idx = blockIdx.x * blockDim.x + threadIdx.x;   // QKV_LD*32
    if (idx >= QKV_LD * 32) return;
    int n = idx >> 5, j = idx & 31;
    o[idx] = (j < 12) ? f2bf(loraB[n * 12 + j]) : (unsigned short)0;
}

// ---------------------------------------------------------------------------
// vT[head][ch][key] = qkv_bf[key][1536 + head*64 + ch]
// ---------------------------------------------------------------------------
__global__ void vT_kernel(const unsigned short* __restrict__ qkvbf,
                          unsigned short* __restrict__ vT) {
    int idx = blockIdx.x * blockDim.x + threadIdx.x;  // 12*64*4096
    int key = idx & 4095;
    int c = (idx >> 12) & 63;
    int head = idx >> 18;
    vT[((size_t)(head * 64 + c) << 12) + key] =
        qkvbf[(size_t)key * QKV_LD + 2 * DIM + head * HEAD_DIM + c];
}

// ---------------------------------------------------------------------------
// GEMM  C[M][N] = A[M][K] @ W[N][K]^T + bias (+ LoRA via one extra WMMA).
// 8 waves / 256 threads per WG: wave w owns rows bx*128 + w*16, all waves
// share one async-staged W slab [64][K] bf16 in LDS (dynamic extern shared).
// ---------------------------------------------------------------------------
template <bool LORA, bool OUTBF>
__global__ void gemm_bf16_kernel(const unsigned short* __restrict__ A,
                                 const unsigned short* __restrict__ W,
                                 const float* __restrict__ bias,
                                 const unsigned short* __restrict__ t32,
                                 const unsigned short* __restrict__ loraB32,
                                 void* __restrict__ Cout,
                                 int M, int N, int K) {
    extern __shared__ unsigned short smem[];   // Wlds [64][K]
    const int tid = threadIdx.x;
    const int wave = tid >> 5;
    const int row0 = blockIdx.x * 128 + wave * 16;
    const int n0   = blockIdx.y * 64;

    // async-stage W[n0..n0+63][0..K) -> LDS (16B per transfer, offset = i*16)
    const int nseg = K >> 3;                   // 16B segments per row
    for (int i = tid; i < 64 * nseg; i += 256) {
        int row = i / nseg, seg = i % nseg;
        async_b128((unsigned)i * 16u, W + (size_t)(n0 + row) * K + seg * 8);
    }

    v8f acc[4] = {};
    if (LORA) {  // rank-12 update as a K=32 WMMA (overlaps async staging)
        v16bf ta = load_a_bf(t32, 32, row0, 0);
#pragma unroll
        for (int j = 0; j < 4; ++j)
            acc[j] = wmma_bf16(ta, load_b_bf(loraB32, 32, n0 + j * 16, 0),
                               acc[j]);
    }
    WAIT_ASYNC_0();
    __syncthreads();

    for (int k0 = 0; k0 < K; k0 += 32) {
        v16bf a = load_a_bf(A, K, row0, k0);
#pragma unroll
        for (int j = 0; j < 4; ++j)
            acc[j] = wmma_bf16(a, load_b_bf(smem, K, j * 16, k0), acc[j]);
    }
    const int nloc = tid & 15, hi = (tid >> 4) & 1;
    float bn[4];
#pragma unroll
    for (int j = 0; j < 4; ++j) bn[j] = bias[n0 + j * 16 + nloc];
#pragma unroll
    for (int r = 0; r < 8; ++r) {
        int m = row0 + r + 8 * hi;
#pragma unroll
        for (int j = 0; j < 4; ++j) {
            int n = n0 + j * 16 + nloc;
            float v = acc[j][r] + bn[j];
            if (OUTBF)
                ((unsigned short*)Cout)[(size_t)m * N + n] = f2bf(v);
            else
                ((float*)Cout)[(size_t)m * N + n] = v;
        }
    }
}

// ---------------------------------------------------------------------------
// Flash attention. 8 waves / 256 threads per WG = one head x 128 query rows;
// wave w owns 16 rows. K/V key-blocks async-staged into double-buffered LDS
// shared by all 8 waves. Rel-pos bias via WMMA from bf16 tables:
//   relH[m][kh] = q[m] . rel_pos_h[qh-kh+63]
//   Ew[m][d]    = q[m] . rel_pos_w[qw0+d], d=0..79
//   relW[m][kw] = Ew[m][m+63-kw]   (kb-invariant -> registers)
//
// Dynamic LDS layout (ushort units):
//   [0,8192)      Klds[2][64][64]
//   [8192,16384)  Vlds[2][64][64]        (vT rows: [ch][key])
//   16384 + wave*6016 : relH f32[16][64] | Ew f32[16][88] | Pl us[16][72]
// ---------------------------------------------------------------------------
__device__ __forceinline__ void stage_kv_flash(const unsigned short* kp,
                                               const unsigned short* vTh,
                                               int key0, int buf, int tid) {
    unsigned kbase = (unsigned)buf * 8192u;            // bytes
    unsigned vbase = 16384u + (unsigned)buf * 8192u;   // bytes
#pragma unroll
    for (int s = 0; s < 2; ++s) {
        int id = s * 256 + tid;                        // 0..511
        int row = id >> 3, seg = id & 7;
        async_b128(kbase + row * 128 + seg * 16,
                   kp + (size_t)(key0 + row) * QKV_LD + seg * 8);
        async_b128(vbase + row * 128 + seg * 16,
                   vTh + ((size_t)row << 12) + key0 + seg * 8);
    }
}

__global__ void flash_attn_kernel(const unsigned short* __restrict__ qkvbf,
                                  const unsigned short* __restrict__ vT,
                                  const unsigned short* __restrict__ rhbf,
                                  const unsigned short* __restrict__ rwbf,
                                  unsigned short* __restrict__ attnbf) {
    extern __shared__ unsigned short smem[];
    const int tid = threadIdx.x;
    const int wave = tid >> 5;
    const int head = blockIdx.y;
    const int row0 = blockIdx.x * 128 + wave * 16;
    const int qh = row0 >> 6;
    const int qw0 = row0 & 63;
    const unsigned short* qp = qkvbf + head * HEAD_DIM;
    const unsigned short* kp = qkvbf + DIM + head * HEAD_DIM;
    const unsigned short* vTh = vT + ((size_t)head * HEAD_DIM << 12);

    unsigned short* warea = smem + 16384 + wave * 6016;
    float* relH = (float*)warea;               // [16][64]
    float* Ew   = (float*)(warea + 2048);      // [16][88]
    unsigned short* Pl = warea + 4864;         // [16][72]

    const int nloc = tid & 15, hi = (tid >> 4) & 1;

    // kick off async staging of key-block 0 (overlaps bias GEMMs below)
    stage_kv_flash(kp, vTh, 0, 0, tid);

    // unscaled q as A operand (reused for S and both bias GEMMs)
    v16bf qa0 = load_a_bf(qp, QKV_LD, row0, 0);
    v16bf qa1 = load_a_bf(qp, QKV_LD, row0, 32);

    // relH = q @ Rh^T   (Rh row = qh - kh + 63, in [0,126]); bf16 table
#pragma unroll
    for (int j = 0; j < 4; ++j) {
        v8f z = {};
        z = wmma_bf16(qa0, load_b_bf_step(rhbf, HEAD_DIM,
                                          qh + 63 - j * 16, -1, 126, 0), z);
        z = wmma_bf16(qa1, load_b_bf_step(rhbf, HEAD_DIM,
                                          qh + 63 - j * 16, -1, 126, 32), z);
#pragma unroll
        for (int r = 0; r < 8; ++r) relH[(r + 8 * hi) * 64 + j * 16 + nloc] = z[r];
    }
    // Ew = q @ Rw_rows^T  (row = qw0 + d, clamped; d=79 never consumed)
#pragma unroll
    for (int j = 0; j < 5; ++j) {
        v8f z = {};
        z = wmma_bf16(qa0, load_b_bf_step(rwbf, HEAD_DIM,
                                          qw0 + j * 16, 1, 126, 0), z);
        z = wmma_bf16(qa1, load_b_bf_step(rwbf, HEAD_DIM,
                                          qw0 + j * 16, 1, 126, 32), z);
#pragma unroll
        for (int r = 0; r < 8; ++r) Ew[(r + 8 * hi) * 88 + j * 16 + nloc] = z[r];
    }
    // per-wave LDS, same-wave in-order DS -> no barrier needed before reads
    float rw[4][8];
#pragma unroll
    for (int j = 0; j < 4; ++j)
#pragma unroll
        for (int r = 0; r < 8; ++r) {
            int m = r + 8 * hi;
            rw[j][r] = Ew[m * 88 + m + 63 - (j * 16 + nloc)];
        }

    float mrow[8], lrow[8];
    v8f o[4] = {};
#pragma unroll
    for (int r = 0; r < 8; ++r) { mrow[r] = -1e30f; lrow[r] = 0.f; }

    for (int kb = 0; kb < 64; ++kb) {          // kb == kh
        if (kb + 1 < 64) {
            stage_kv_flash(kp, vTh, (kb + 1) * 64, (kb + 1) & 1, tid);
            WAIT_ASYNC_4();                    // wait for block kb's 4 copies
        } else {
            WAIT_ASYNC_0();
        }
        __syncthreads();
        const unsigned short* Kb = smem + (kb & 1) * 4096;
        const unsigned short* Vb = smem + 8192 + (kb & 1) * 4096;

        // S = q @ k^T (4 tiles of 16x16, K=64 as 2x32) from shared LDS
        v8f s[4];
#pragma unroll
        for (int j = 0; j < 4; ++j) {
            v8f z = {};
            z = wmma_bf16(qa0, load_b_bf(Kb, 64, j * 16, 0), z);
            z = wmma_bf16(qa1, load_b_bf(Kb, 64, j * 16, 32), z);
            s[j] = z;
        }
        // scale + rel-pos bias
        float rh[8];
#pragma unroll
        for (int r = 0; r < 8; ++r) rh[r] = relH[(r + 8 * hi) * 64 + kb];
#pragma unroll
        for (int j = 0; j < 4; ++j)
#pragma unroll
            for (int r = 0; r < 8; ++r)
                s[j][r] = s[j][r] * SCALE + (rh[r] + rw[j][r]);
        // online softmax (rows span 16 lanes of each half-wave)
#pragma unroll
        for (int r = 0; r < 8; ++r) {
            float mx = fmaxf(fmaxf(s[0][r], s[1][r]), fmaxf(s[2][r], s[3][r]));
#pragma unroll
            for (int off = 8; off; off >>= 1)
                mx = fmaxf(mx, __shfl_xor(mx, off, 32));
            float mnew = fmaxf(mrow[r], mx);
            float alpha = __expf(mrow[r] - mnew);
            mrow[r] = mnew;
            float psum = 0.f;
#pragma unroll
            for (int j = 0; j < 4; ++j) {
                float pv = __expf(s[j][r] - mnew);
                s[j][r] = pv;
                psum += pv;
            }
#pragma unroll
            for (int off = 8; off; off >>= 1)
                psum += __shfl_xor(psum, off, 32);
            lrow[r] = lrow[r] * alpha + psum;
#pragma unroll
            for (int j = 0; j < 4; ++j) o[j][r] *= alpha;
        }
        // P: C-layout -> per-wave LDS -> A-layout (same-wave, in-order DS)
#pragma unroll
        for (int j = 0; j < 4; ++j)
#pragma unroll
            for (int r = 0; r < 8; ++r)
                Pl[(r + 8 * hi) * 72 + j * 16 + nloc] = f2bf(s[j][r]);
        v16bf pa0 = load_a_bf(Pl, 72, 0, 0);
        v16bf pa1 = load_a_bf(Pl, 72, 0, 32);
        // O += P @ V   (B operand = contiguous rows of vT[ch][key] in LDS)
#pragma unroll
        for (int j = 0; j < 4; ++j) {
            o[j] = wmma_bf16(pa0, load_b_bf(Vb, 64, j * 16, 0), o[j]);
            o[j] = wmma_bf16(pa1, load_b_bf(Vb, 64, j * 16, 32), o[j]);
        }
        __syncthreads();
    }
    // normalize + write bf16 attn output at [tok][head*64+ch]
    float inv[8];
#pragma unroll
    for (int r = 0; r < 8; ++r) inv[r] = 1.f / lrow[r];
#pragma unroll
    for (int j = 0; j < 4; ++j)
#pragma unroll
        for (int r = 0; r < 8; ++r) {
            int m = row0 + r + 8 * hi;
            attnbf[(size_t)m * DIM + head * HEAD_DIM + j * 16 + nloc] =
                f2bf(o[j][r] * inv[r]);
        }
}

// ---------------------------------------------------------------------------
extern "C" void kernel_launch(void* const* d_in, const int* in_sizes, int n_in,
                              void* d_out, int out_size, void* d_ws,
                              size_t ws_size, hipStream_t stream) {
    const float* x         = (const float*)d_in[0];
    const float* w_qkv     = (const float*)d_in[1];
    const float* b_qkv     = (const float*)d_in[2];
    const float* lora_A    = (const float*)d_in[3];
    const float* lora_B    = (const float*)d_in[4];
    const float* w_proj    = (const float*)d_in[5];
    const float* b_proj    = (const float*)d_in[6];
    const float* rel_pos_h = (const float*)d_in[7];
    const float* rel_pos_w = (const float*)d_in[8];
    float* out = (float*)d_out;

    char* ws = (char*)d_ws;
    size_t off = 0;
    unsigned short* t32     = (unsigned short*)(ws + off); off += (size_t)NTOK * 32 * 2;
    unsigned short* loraB32 = (unsigned short*)(ws + off); off += (size_t)QKV_LD * 32 * 2;
    unsigned short* rhbf    = (unsigned short*)(ws + off); off += 16384;   // 127*64 used
    unsigned short* rwbf    = (unsigned short*)(ws + off); off += 16384;
    unsigned short* xbf     = (unsigned short*)(ws + off); off += (size_t)NTOK * DIM * 2;
    unsigned short* wqkvbf  = (unsigned short*)(ws + off); off += (size_t)QKV_LD * DIM * 2;
    unsigned short* wprojbf = (unsigned short*)(ws + off); off += (size_t)DIM * DIM * 2;
    unsigned short* qkvbf   = (unsigned short*)(ws + off); off += (size_t)NTOK * QKV_LD * 2;
    unsigned short* vTbf    = (unsigned short*)(ws + off); off += (size_t)NUM_HEADS * HEAD_DIM * NTOK * 2;
    unsigned short* attnbf  = (unsigned short*)(ws + off); off += (size_t)NTOK * DIM * 2;

    // 1) one-shot fp32 -> bf16 conversions (inputs, weights, rel-pos tables)
    {
        int np = NTOK * DIM / 2;
        cvt_bf16_kernel<<<(np + 255) / 256, 256, 0, stream>>>(x, (unsigned*)xbf, np);
        np = QKV_LD * DIM / 2;
        cvt_bf16_kernel<<<(np + 255) / 256, 256, 0, stream>>>(w_qkv, (unsigned*)wqkvbf, np);
        np = DIM * DIM / 2;
        cvt_bf16_kernel<<<(np + 255) / 256, 256, 0, stream>>>(w_proj, (unsigned*)wprojbf, np);
        np = 127 * HEAD_DIM / 2;
        cvt_bf16_kernel<<<(np + 255) / 256, 256, 0, stream>>>(rel_pos_h, (unsigned*)rhbf, np);
        cvt_bf16_kernel<<<(np + 255) / 256, 256, 0, stream>>>(rel_pos_w, (unsigned*)rwbf, np);
    }
    // 2) LoRA operands, zero-padded to K=32 (bf16)
    lora_t_kernel<<<(NTOK * 32 + 255) / 256, 256, 0, stream>>>(x, lora_A, t32);
    loraB32_kernel<<<(QKV_LD * 32 + 255) / 256, 256, 0, stream>>>(lora_B, loraB32);

    // 3) QKV projection + bias + LoRA (WMMA) -> bf16 qkv
    {
        dim3 grid(NTOK / 128, QKV_LD / 64);
        gemm_bf16_kernel<true, true><<<grid, 256, 64 * DIM * 2, stream>>>(
            xbf, wqkvbf, b_qkv, t32, loraB32, qkvbf, NTOK, QKV_LD, DIM);
    }
    // 4) per-head V transpose (bf16, L2-resident)
    {
        int n = NUM_HEADS * HEAD_DIM * NTOK;
        vT_kernel<<<(n + 255) / 256, 256, 0, stream>>>(qkvbf, vTbf);
    }
    // 5) flash attention with async-staged K/V and WMMA rel-pos bias
    {
        dim3 grid(NTOK / 128, NUM_HEADS);
        size_t lds = (16384 + 8 * 6016) * sizeof(unsigned short);  // 129 KB
        flash_attn_kernel<<<grid, 256, lds, stream>>>(qkvbf, vTbf, rhbf, rwbf,
                                                      attnbf);
    }
    // 6) output projection -> fp32 out
    {
        dim3 grid(NTOK / 128, DIM / 64);
        gemm_bf16_kernel<false, false><<<grid, 256, 64 * DIM * 2, stream>>>(
            attnbf, wprojbf, b_proj, nullptr, nullptr, out, NTOK, DIM, DIM);
    }
}
